// SegmentationLoss_30812095382005
// MI455X (gfx1250) — 8-bit matrix operand order repaired and checked by reading, compiled, not run on hardware
//
#include <hip/hip_runtime.h>


#define BQ      32
#define NCH     4
#define HWPIX   (512 * 512)
#define BLOCK   256
#define WAVES   8
#define CHUNKS  64
#define PIX_PER_BLOCK (HWPIX / CHUNKS)
#define PIX_PER_WAVE  (PIX_PER_BLOCK / WAVES)
#define TILES_PER_WAVE (PIX_PER_WAVE / 64)

typedef int v8i __attribute__((ext_vector_type(8)));

__device__ __forceinline__ float bce_term(float p, float t) {
    float e = __builtin_amdgcn_exp2f(fabsf(p) * -1.44269504088896340736f);
    float l = __builtin_amdgcn_logf(1.0f + e);
    return fmaxf(p, 0.0f) - p * t + 0.69314718055994530942f * l;
}

__global__ void seg_init(double* bceSum, unsigned int* conf) {
    if (threadIdx.x == 0) *bceSum = 0.0;
    unsigned i = threadIdx.x;
    if (i < BQ * 16) conf[i] = 0u;
}

__global__ __launch_bounds__(BLOCK) void seg_main(const float* __restrict__ P,
                                                  const float* __restrict__ T,
                                                  double* __restrict__ bceSum,
                                                  unsigned int* __restrict__ conf) {
    __shared__ __align__(16) unsigned char sPred[WAVES][64];
    __shared__ __align__(16) unsigned char sTgt [WAVES][64];
    __shared__ float sRed[BLOCK];

    const int tid  = threadIdx.x;
    const int lane = tid & 31;
    const int wv   = tid >> 5;
    const int b    = blockIdx.x / CHUNKS;
    const int chnk = blockIdx.x % CHUNKS;

    const float* Pb = P + (size_t)b * (NCH * (size_t)HWPIX);
    const float* Tb = T + (size_t)b * (NCH * (size_t)HWPIX);
    const int waveBase = chnk * PIX_PER_BLOCK + wv * PIX_PER_WAVE;

    float bce = 0.0f;
    v8i acc = {};

    const int grp = (lane >> 4) & 1;
    const unsigned mc = 0x01010101u * (unsigned)(lane & 3);

    unsigned char* pb = &sPred[wv][0];
    unsigned char* tb = &sTgt[wv][0];

    for (int t = 0; t < TILES_PER_WAVE; ++t) {
        const int pix = waveBase + t * 64 + 2 * lane;
        float2 pv[NCH], tv[NCH];
        #pragma unroll
        for (int c = 0; c < NCH; ++c) {
            pv[c] = *(const float2*)(Pb + (size_t)c * HWPIX + pix);
            tv[c] = *(const float2*)(Tb + (size_t)c * HWPIX + pix);
        }

        float bp0 = pv[0].x, bp1 = pv[0].y, bt0 = tv[0].x, bt1 = tv[0].y;
        int cp0 = 0, cp1 = 0, ct0 = 0, ct1 = 0;
        #pragma unroll
        for (int c = 1; c < NCH; ++c) {
            if (pv[c].x > bp0) { bp0 = pv[c].x; cp0 = c; }
            if (pv[c].y > bp1) { bp1 = pv[c].y; cp1 = c; }
            if (tv[c].x > bt0) { bt0 = tv[c].x; ct0 = c; }
            if (tv[c].y > bt1) { bt1 = tv[c].y; ct1 = c; }
        }

        #pragma unroll
        for (int c = 0; c < NCH; ++c) {
            bce += bce_term(pv[c].x, tv[c].x);
            bce += bce_term(pv[c].y, tv[c].y);
        }

        *(unsigned short*)(pb + 2 * lane) = (unsigned short)(cp0 | (cp1 << 8));
        *(unsigned short*)(tb + 2 * lane) = (unsigned short)(ct0 | (ct1 << 8));
        __asm__ volatile("" ::: "memory");

        unsigned aw[8];
        #pragma unroll
        for (int j = 0; j < 4; ++j) {
            uint2 v = *(const uint2*)(pb + j * 16 + grp * 8);
            aw[2 * j] = v.x; aw[2 * j + 1] = v.y;
        }
        unsigned bw[8];
        #pragma unroll
        for (int j = 0; j < 4; ++j) {
            uint2 v = *(const uint2*)(tb + j * 16 + grp * 8);
            bw[2 * j] = v.x; bw[2 * j + 1] = v.y;
        }
        __asm__ volatile("" ::: "memory");

        v8i A, Bm;
        #pragma unroll
        for (int w = 0; w < 8; ++w) {
            unsigned x = aw[w] ^ mc;
            unsigned s = x | (x >> 1);
            A[w] = (int)(~s & 0x01010101u);
            x = bw[w] ^ mc;
            s = x | (x >> 1);
            Bm[w] = (int)(~s & 0x01010101u);
        }

        acc = __builtin_amdgcn_wmma_i32_16x16x64_iu8(false, A, false, Bm, acc,
                                                     false, false);
    }

    if (lane < 4) {
        atomicAdd(&conf[(b << 4) + 0  + lane], (unsigned)acc[0]);
        atomicAdd(&conf[(b << 4) + 4  + lane], (unsigned)acc[1]);
        atomicAdd(&conf[(b << 4) + 8  + lane], (unsigned)acc[2]);
        atomicAdd(&conf[(b << 4) + 12 + lane], (unsigned)acc[3]);
    }

    sRed[tid] = bce;
    __syncthreads();
    #pragma unroll
    for (int s = BLOCK / 2; s > 0; s >>= 1) {
        if (tid < s) sRed[tid] += sRed[tid + s];
        __syncthreads();
    }
    if (tid == 0) atomicAdd(bceSum, (double)sRed[0]);
}

__global__ void seg_final(const double* __restrict__ bceSum,
                          const unsigned int* __restrict__ conf,
                          float* __restrict__ out) {
    __shared__ float sv[32];
    const int b = threadIdx.x;
    float r = 0.0f;
    if (b < BQ) {
        unsigned c[16];
        #pragma unroll
        for (int i = 0; i < 16; ++i) c[i] = conf[b * 16 + i];
        float ious = 0.0f;
        int present = 0;
        #pragma unroll
        for (int k = 0; k < 4; ++k) {
            unsigned tp = c[k * 4 + k];
            unsigned pc = c[k * 4 + 0] + c[k * 4 + 1] + c[k * 4 + 2] + c[k * 4 + 3];
            unsigned tc = c[0 * 4 + k] + c[1 * 4 + k] + c[2 * 4 + k] + c[3 * 4 + k];
            unsigned den = pc + tc - tp;
            if (den > 0u) ious += (float)tp / (float)den;
            if (tc > 0u) present++;
        }
        r = (present > 0) ? ious / (float)present : 0.0f;
    }
    sv[threadIdx.x] = r;
    __syncthreads();
    if (threadIdx.x == 0) {
        float s = 0.0f;
        #pragma unroll
        for (int i = 0; i < 32; ++i) s += sv[i];
        out[1] = 1.0f - s / (float)BQ;
        out[0] = (float)(*bceSum / ((double)BQ * NCH * (double)HWPIX));
    }
}

extern "C" void kernel_launch(void* const* d_in, const int* in_sizes, int n_in,
                              void* d_out, int out_size, void* d_ws, size_t ws_size,
                              hipStream_t stream) {
    const float* P = (const float*)d_in[0];
    const float* T = (const float*)d_in[1];
    float* out = (float*)d_out;

    double* bceSum     = (double*)d_ws;
    unsigned int* conf = (unsigned int*)((char*)d_ws + 16);

    seg_init <<<1, 512, 0, stream>>>(bceSum, conf);
    seg_main <<<BQ * CHUNKS, BLOCK, 0, stream>>>(P, T, bceSum, conf);
    seg_final<<<1, 32, 0, stream>>>(bceSum, conf, out);
    (void)in_sizes; (void)n_in; (void)out_size; (void)ws_size;
}
